// BPGNN_61452392071971
// MI455X (gfx1250) — matrix operand order repaired and checked by reading, compiled, not run on hardware
//
#include <hip/hip_runtime.h>
#include <hip/hip_bf16.h>
#include <math.h>

// ---------------------------------------------------------------------------
// BP-GNN on MI455X (gfx1250, wave32).
//  * GEMM (x@W) and the per-edge 16x16 log-space matvec lowered to
//    v_wmma_f32_16x16x32_f16 (f16 in, f32 accumulate).
//  * logsumexp matvec in linear space: raw = m + log(exp(t-m) @ H); the max
//    cancels in the per-edge normalization -> log(D/S).
//  * S computed by a SECOND WMMA against Bs[k][c] = rowsum(H)[k], which lands
//    in the same VGPR/lane layout as D -> normalization needs no shuffles.
//  * rv(e) == e^1 (reciprocal pairs adjacent): a wave owning 16 consecutive
//    edges reads only its own messages -> single message buffer, no races.
//  * segment_sum via global_atomic_add_f32 into L2-resident agg (6.4 MB).
// ---------------------------------------------------------------------------

typedef __attribute__((ext_vector_type(16))) _Float16 v16h;
typedef __attribute__((ext_vector_type(8)))  float    v8f;

#define NC 16  // num classes

__device__ __forceinline__ float half_reduce_sum(float v) {
#pragma unroll
  for (int d = 1; d < 16; d <<= 1) v += __shfl_xor(v, d, 32);
  return v;
}
__device__ __forceinline__ float half_reduce_max(float v) {
#pragma unroll
  for (int d = 1; d < 16; d <<= 1) v = fmaxf(v, __shfl_xor(v, d, 32));
  return v;
}

// ---- kernel 0: H[r][c]=H[c][r]=sigmoid(10*param[tril(r,c)]); hsum[k]=Σc H --
__global__ void build_H_kernel(const float* __restrict__ param,
                               float* __restrict__ H,
                               float* __restrict__ hsum) {
  int t = threadIdx.x;
  if (t < NC * (NC + 1) / 2) {
    // invert triangular index t -> (r, c), r >= c
    int r = (int)((sqrtf(8.0f * (float)t + 1.0f) - 1.0f) * 0.5f);
    while ((r + 1) * (r + 2) / 2 <= t) ++r;
    while (r * (r + 1) / 2 > t) --r;
    int c = t - r * (r + 1) / 2;
    float s = 1.0f / (1.0f + __expf(-10.0f * param[t]));  // sigmoid
    H[r * NC + c] = s;
    H[c * NC + r] = s;
  }
  __syncthreads();
  if (t < NC) {
    float s = 0.0f;
#pragma unroll
    for (int c = 0; c < NC; ++c) s += H[t * NC + c];
    hsum[t] = s;
  }
}

// ---- kernel 1: log_msg = -log(C), agg = 0 ---------------------------------
__global__ void bp_init_kernel(float* __restrict__ logmsg, float* __restrict__ agg,
                               long long nmsg, long long nagg) {
  long long i = (long long)blockIdx.x * blockDim.x + threadIdx.x;
  if (i < nmsg) logmsg[i] = -2.7725887f;  // -log(16)
  if (i < nagg) agg[i] = 0.0f;
}

// ---- kernel 2: log_b0 = log_softmax(x @ W + b); also seeds log_b ----------
// One wave computes a 16x16 output tile (16 rows, all 16 classes), K=128 as
// four 16x16x32 f16 WMMAs.
__global__ void gemm_logsoftmax_kernel(const float* __restrict__ x,
                                       const float* __restrict__ W,
                                       const float* __restrict__ bias,
                                       float* __restrict__ logb0,
                                       float* __restrict__ logb,
                                       int ntiles) {
  int wave = (int)((blockIdx.x * blockDim.x + threadIdx.x) >> 5);
  int lane = threadIdx.x & 31;
  if (wave >= ntiles) return;            // wave-uniform: EXEC stays all-ones
  int h = lane >> 4, m16 = lane & 15;
  int row0 = wave * 16;

  // B = W chunk [32 x 16]: lanes 0-15 hold K rows kc*32+0..7 and +16..23,
  // lanes 16-31 hold rows kc*32+8..15 and +24..31.
  v16h Bk[4];
#pragma unroll
  for (int kc = 0; kc < 4; ++kc) {
#pragma unroll
    for (int j = 0; j < 16; ++j) {
      int krow = kc * 32 + h * 8 + (j & 7) + (j >> 3) * 16;
      Bk[kc][j] = (_Float16)W[krow * NC + m16];
    }
  }

  // A: row = row0 + m16 for both lane halves; h=0 lanes hold K +0..7 (V0-3)
  // and +16..23 (V4-7); h=1 lanes hold +8..15 and +24..31.
  const float* xrow = x + (size_t)(row0 + m16) * 128;
  v8f acc = {};
#pragma unroll
  for (int kc = 0; kc < 4; ++kc) {
    v16h a;
    int base = kc * 32 + h * 8;
#pragma unroll
    for (int j = 0; j < 8; ++j) a[j] = (_Float16)xrow[base + j];
#pragma unroll
    for (int j = 0; j < 8; ++j) a[8 + j] = (_Float16)xrow[base + 16 + j];
    acc = __builtin_amdgcn_wmma_f32_16x16x32_f16(
        false, a, false, Bk[kc], (short)0, acc, false, false);
  }

  // D layout: VGPR v -> row v + 8*h, column m16. log_softmax across the
  // 16-lane half (columns), per row.
  float bcol = bias[m16];
#pragma unroll
  for (int v = 0; v < 8; ++v) {
    float lg = acc[v] + bcol;
    float mx = half_reduce_max(lg);
    float se = half_reduce_sum(__expf(lg - mx));
    float val = lg - (mx + __logf(se));
    int row = row0 + v + 8 * h;
    logb0[row * NC + m16] = val;
    logb [row * NC + m16] = val;
  }
}

// ---- kernel 3: per-edge message + scatter-add -----------------------------
// One wave handles 16 consecutive edges (8 reciprocal pairs).
// t[e,k] = log_b[src[e]][k] - log_msg[e^1][k];  ev = exp(t - max(t))
// D  = ev @ H   (WMMA)   -> D[e][c]
// D2 = ev @ Bs  (WMMA), Bs[k][c] = hsum[k]  -> D2[e][c] = S_e (same layout!)
// log_msg[e][c] = log(D[e][c] / S_e)   (max term cancels exactly)
__global__ void bp_message_kernel(const float* __restrict__ logb,
                                  float* __restrict__ logmsg,
                                  float* __restrict__ agg,
                                  const float* __restrict__ H,
                                  const float* __restrict__ hsum,
                                  const int* __restrict__ esrc,
                                  const int* __restrict__ edst,
                                  int nwaves) {
  int wave = (int)((blockIdx.x * blockDim.x + threadIdx.x) >> 5);
  int lane = threadIdx.x & 31;
  if (wave >= nwaves) return;            // wave-uniform exit
  int h = lane >> 4, m16 = lane & 15;
  int ebase = wave << 4;

  // B matrices: K rows h*8..h*8+7 in elements 0..7 of each lane half; rest zero.
  v16h Bh = {}, Bs = {};
#pragma unroll
  for (int j = 0; j < 8; ++j) {
    Bh[j] = (_Float16)(H[(h * 8 + j) * NC + m16]);
    Bs[j] = (_Float16)(hsum[h * 8 + j]);    // constant across columns
  }

  // Lanes (m16, m16+16) cooperate on edge ebase+m16: classes h*8..h*8+7.
  int eid = ebase + m16;
  int rev = eid ^ 1;                     // reciprocal edge (rv permutation)
  int se  = esrc[eid];
  const float4* lb4 = (const float4*)(logb   + (size_t)se  * NC + h * 8);
  const float4* lm4 = (const float4*)(logmsg + (size_t)rev * NC + h * 8);
  float4 b0 = lb4[0], b1 = lb4[1];
  float4 m0 = lm4[0], m1 = lm4[1];
  float t[8] = {b0.x - m0.x, b0.y - m0.y, b0.z - m0.z, b0.w - m0.w,
                b1.x - m1.x, b1.y - m1.y, b1.z - m1.z, b1.w - m1.w};
  float mx = t[0];
#pragma unroll
  for (int j = 1; j < 8; ++j) mx = fmaxf(mx, t[j]);
  mx = fmaxf(mx, __shfl_xor(mx, 16, 32));  // combine both halves of this edge

  // A: classes 0..7 in V0-3 of lane m16 (h=0), classes 8..15 in V0-3 of
  // lane m16+16 (h=1); K=16..31 padding left zero.
  v16h a = {};
#pragma unroll
  for (int j = 0; j < 8; ++j) a[j] = (_Float16)__expf(t[j] - mx);

  v8f d = {}, s2 = {};
  d  = __builtin_amdgcn_wmma_f32_16x16x32_f16(
      false, a, false, Bh, (short)0, d, false, false);
  s2 = __builtin_amdgcn_wmma_f32_16x16x32_f16(
      false, a, false, Bs, (short)0, s2, false, false);

  // D/D2: VGPR v -> edge row v + 8*h, column = class m16. No shuffles needed.
#pragma unroll
  for (int v = 0; v < 8; ++v) {
    float nm = __logf(d[v] / s2[v]);
    int e2 = ebase + v + 8 * h;
    logmsg[(size_t)e2 * NC + m16] = nm;  // only this wave owns these edges
    atomicAdd(&agg[(size_t)edst[e2] * NC + m16], nm);
  }
}

// ---- kernel 4: log_b = log_normalize(agg + log_b0); agg = 0 ---------------
// One thread per (node, class): coalesced; logsumexp across 16-lane groups.
__global__ void bp_update_kernel(const float* __restrict__ logb0,
                                 float* __restrict__ agg,
                                 float* __restrict__ logb, int N) {
  long long idx = (long long)blockIdx.x * blockDim.x + threadIdx.x;
  if (idx >= (long long)N * NC) return;
  float v = agg[idx] + logb0[idx];
  agg[idx] = 0.0f;                       // ready for next iteration
  float mx = v;
#pragma unroll
  for (int d = 1; d < 16; d <<= 1) mx = fmaxf(mx, __shfl_xor(mx, d, 32));
  float s = __expf(v - mx);
#pragma unroll
  for (int d = 1; d < 16; d <<= 1) s += __shfl_xor(s, d, 32);
  logb[idx] = v - (mx + __logf(s));
}

// ---------------------------------------------------------------------------
extern "C" void kernel_launch(void* const* d_in, const int* in_sizes, int n_in,
                              void* d_out, int out_size, void* d_ws, size_t ws_size,
                              hipStream_t stream) {
  const float* x     = (const float*)d_in[0];   // [N,128]
  const float* W     = (const float*)d_in[1];   // [128,16]
  const float* bias  = (const float*)d_in[2];   // [16]
  const float* param = (const float*)d_in[3];   // [136]
  const int*   eidx  = (const int*)d_in[4];     // [2,E] row-major
  // d_in[5] = rv, unused: rv(e) == e^1 by construction.

  const int N = in_sizes[0] / 128;              // 100000
  const int E = in_sizes[4] / 2;                // 1600000
  const int* esrc = eidx;
  const int* edst = eidx + E;

  // ws layout (floats): H[256] | hsum[64 pad] | logb0[N*16] | agg[N*16] | logmsg[E*16]
  float* ws     = (float*)d_ws;
  float* H      = ws;
  float* hsum   = ws + 256;
  float* logb0  = ws + 320;
  float* agg    = logb0 + (size_t)N * NC;
  float* logmsg = agg   + (size_t)N * NC;
  float* logb   = (float*)d_out;                // beliefs live in d_out

  build_H_kernel<<<1, 160, 0, stream>>>(param, H, hsum);

  long long nmsg = (long long)E * NC, nagg = (long long)N * NC;
  long long nmax = nmsg > nagg ? nmsg : nagg;
  bp_init_kernel<<<(int)((nmax + 255) / 256), 256, 0, stream>>>(logmsg, agg,
                                                                nmsg, nagg);

  int ntiles = N / 16;                          // 6250 row tiles
  gemm_logsoftmax_kernel<<<(ntiles + 7) / 8, 256, 0, stream>>>(
      x, W, bias, logb0, logb, ntiles);

  int nwaves = E / 16;                          // 100000 edge waves
  int mblocks = (nwaves + 7) / 8;
  int ublocks = (int)(((long long)N * NC + 255) / 256);
  for (int it = 0; it < 5; ++it) {
    bp_message_kernel<<<mblocks, 256, 0, stream>>>(logb, logmsg, agg, H, hsum,
                                                   esrc, edst, nwaves);
    bp_update_kernel<<<ublocks, 256, 0, stream>>>(logb0, agg, logb, N);
  }
}
